// SelfAttention_84722524881009
// MI455X (gfx1250) — matrix operand order repaired and checked by reading, not compiled, not run on hardware
//
#include <hip/hip_runtime.h>

#define HID 1024
#define SEQ 2048
#define NBATCH 4
#define NHEAD 16
#define DHEAD 64
#define NTOK (NBATCH * SEQ)

typedef __attribute__((ext_vector_type(16))) __bf16 v16bf;
typedef __attribute__((ext_vector_type(8)))  __bf16 v8bf;
typedef __attribute__((ext_vector_type(8)))  float  v8f;

static __device__ __forceinline__ __bf16 f2bf(float f) {
  unsigned u = __builtin_bit_cast(unsigned, f);
  unsigned r = u + 0x7FFFu + ((u >> 16) & 1u);   // round-to-nearest-even
  unsigned short h = (unsigned short)(r >> 16);
  return __builtin_bit_cast(__bf16, h);
}

static __device__ __forceinline__ v16bf make16(v8bf lo, v8bf hi) {
  v16bf a;
#pragma unroll
  for (int i = 0; i < 8; ++i) { a[i] = lo[i]; a[i + 8] = hi[i]; }
  return a;
}

// A fragment, 16x32 bf16 (M x K). ISA layout: lanes 0-15 hold rows 0-15 with
// K = {0..7, 16..23}; lanes 16-31 same rows with K = {8..15, 24..31}.
static __device__ __forceinline__ v16bf load_a(const __bf16* base, int row0, int ld,
                                               int k0, int lane) {
  int r  = row0 + (lane & 15);
  int kb = k0 + ((lane >> 4) << 3);
  const __bf16* p = base + (size_t)r * ld + kb;
  v8bf lo = *(const v8bf*)(p);
  v8bf hi = *(const v8bf*)(p + 16);
  return make16(lo, hi);
}

// B fragment, 32x16 bf16 (K x N), from a K-contiguous layout base[n*ld + k].
// ISA layout: lanes 0-15 hold column n with K = {0..7, 16..23}; lanes 16-31 K = {8..15, 24..31}.
static __device__ __forceinline__ v16bf load_b(const __bf16* base, int col0, int ld,
                                               int k0, int lane) {
  int n  = col0 + (lane & 15);
  int kb = k0 + ((lane >> 4) << 3);
  const __bf16* p = base + (size_t)n * ld + kb;
  v8bf lo = *(const v8bf*)(p);
  v8bf hi = *(const v8bf*)(p + 16);
  return make16(lo, hi);
}

static __device__ __forceinline__ v8f wmma_bf16(v16bf a, v16bf b, v8f c) {
  return __builtin_amdgcn_wmma_f32_16x16x32_bf16(false, a, false, b, (short)0, c,
                                                 false, false);
}

// Truncate a generic pointer to a shared variable -> wave-relative LDS byte
// offset (generic LDS addresses carry the aperture in [63:32], offset in [31:0]).
static __device__ __forceinline__ unsigned lds_off(const void* p) {
  return (unsigned)(unsigned long long)p;
}

// One GLOBAL_LOAD_ASYNC_TO_LDS_B128: per-lane 16B global->LDS, ASYNCcnt-tracked.
static __device__ __forceinline__ void async_copy16(unsigned ldsAddr, const void* gptr) {
  unsigned long long ga = (unsigned long long)gptr;
  asm volatile("global_load_async_to_lds_b128 %0, %1, off"
               :: "v"(ldsAddr), "v"(ga) : "memory");
}

static __device__ __forceinline__ void wait_async_all() {
  asm volatile("s_wait_asynccnt 0x0" ::: "memory");
}
// Async "done" returns are in-order per wave; each GEMM stage issues exactly 12
// ops per wave, so <=12 outstanding means the previous stage has fully landed.
static __device__ __forceinline__ void wait_async_prev_stage() {
  asm volatile("s_wait_asynccnt 0xc" ::: "memory");
}

// ---------------------------------------------------------------------------
// Cooperative async staging of one GEMM k-stage (BK=64):
//   A tile 256x64 bf16 (32KB, 8 chunks/thread), B tile 128x64 bf16 (16KB, 4/thread)
static __device__ __forceinline__ void gemm_stage_async(
    const __bf16* __restrict__ ga_base, const __bf16* __restrict__ gb_base,
    int mBlk, int nBlk, int k0, unsigned asOff, unsigned bsOff, int tid) {
#pragma unroll
  for (int j = 0; j < 8; ++j) {
    int c = j * 256 + tid;            // 2048 chunks
    int row = c >> 3, col = (c & 7) << 3;
    async_copy16(asOff + (unsigned)(row * 64 + col) * 2u,
                 ga_base + (size_t)(mBlk + row) * HID + k0 + col);
  }
#pragma unroll
  for (int j = 0; j < 4; ++j) {
    int c = j * 256 + tid;            // 1024 chunks
    int row = c >> 3, col = (c & 7) << 3;
    async_copy16(bsOff + (unsigned)(row * 64 + col) * 2u,
                 gb_base + (size_t)(nBlk + row) * HID + k0 + col);
  }
}

// Two WMMA k-steps out of the staged LDS tiles.
static __device__ __forceinline__ void gemm_compute_stage(
    const __bf16* As, const __bf16* Bs, int wm, int wn, int lane, v8f (&acc)[4][4]) {
#pragma unroll
  for (int kk = 0; kk < 64; kk += 32) {
    v16bf af[4];
#pragma unroll
    for (int jm = 0; jm < 4; ++jm) af[jm] = load_a(As, wm * 64 + 16 * jm, 64, kk, lane);
#pragma unroll
    for (int jn = 0; jn < 4; ++jn) {
      v16bf bfr = load_b(Bs, wn * 64 + 16 * jn, 64, kk, lane);
#pragma unroll
      for (int jm = 0; jm < 4; ++jm) acc[jm][jn] = wmma_bf16(af[jm], bfr, acc[jm][jn]);
    }
  }
}

// Double-buffered async GEMM mainloop. Buffer selection is COMPILE-TIME
// (explicit buf0/buf1 phases) so LDS address-space inference survives and the
// fragment reads stay ds_load_b128 rather than decaying to flat.
static __device__ __forceinline__ void gemm_mainloop(
    const __bf16* __restrict__ ga, const __bf16* __restrict__ gb,
    int mBlk, int nBlk, __bf16 (&As)[2][256 * 64], __bf16 (&Bs)[2][128 * 64],
    int tid, int wm, int wn, int lane, v8f (&acc)[4][4]) {
  const unsigned asOff0 = lds_off(&As[0][0]), asOff1 = lds_off(&As[1][0]);
  const unsigned bsOff0 = lds_off(&Bs[0][0]), bsOff1 = lds_off(&Bs[1][0]);

  gemm_stage_async(ga, gb, mBlk, nBlk, 0, asOff0, bsOff0, tid);
  for (int k0 = 0; k0 < HID; k0 += 128) {   // two 64-wide stages per iteration
    // phase 0: prefetch (k0+64) -> buf1, compute buf0
    if (k0 + 64 < HID) {
      gemm_stage_async(ga, gb, mBlk, nBlk, k0 + 64, asOff1, bsOff1, tid);
      wait_async_prev_stage();
    } else {
      wait_async_all();
    }
    __syncthreads();
    gemm_compute_stage(As[0], Bs[0], wm, wn, lane, acc);
    __syncthreads();
    // phase 1: prefetch (k0+128) -> buf0, compute buf1
    if (k0 + 128 < HID) {
      gemm_stage_async(ga, gb, mBlk, nBlk, k0 + 128, asOff0, bsOff0, tid);
      wait_async_prev_stage();
    } else {
      wait_async_all();
    }
    __syncthreads();
    gemm_compute_stage(As[1], Bs[1], wm, wn, lane, acc);
    __syncthreads();
  }
}

// ---------------------------------------------------------------------------
// fp32 -> bf16 conversion
__global__ void cvt_bf16_kernel(const float* __restrict__ in, __bf16* __restrict__ out,
                                int n) {
  int i = blockIdx.x * blockDim.x + threadIdx.x;
  if (i < n) out[i] = f2bf(in[i]);
}

// W[k][n] (row-major) -> Wt[n][k] bf16 (so GEMM B fragments are K-contiguous)
__global__ void transpose_bf16_kernel(const float* __restrict__ w,
                                      __bf16* __restrict__ wt) {
  int i = blockIdx.x * blockDim.x + threadIdx.x;  // i = n*HID + k
  int n = i >> 10, k = i & (HID - 1);
  wt[i] = f2bf(w[(size_t)k * HID + n]);
}

// ---------------------------------------------------------------------------
// Fused QKV projection + bias + RoPE. blockIdx.z selects Q/K/V.
// Wave tile 64x64 (4x4 fragments), block = 8 waves -> 256x128 block tile.
// A/B tiles double-buffered through LDS with GLOBAL_LOAD_ASYNC_TO_LDS_B128.
__global__ __launch_bounds__(256) void qkv_rope_kernel(
    const __bf16* __restrict__ xb,
    const __bf16* __restrict__ wtq, const __bf16* __restrict__ wtk,
    const __bf16* __restrict__ wtv,
    const float* __restrict__ bq, const float* __restrict__ bk,
    const float* __restrict__ bv,
    __bf16* __restrict__ qout, __bf16* __restrict__ kout,
    __bf16* __restrict__ vtout) {
  __shared__ __bf16 As[2][256 * 64];
  __shared__ __bf16 Bs[2][128 * 64];
  const int tid = threadIdx.x;
  const int lane = tid & 31;
  const int wave = tid >> 5;
  const int wm = wave >> 1, wn = wave & 1;
  const int mode = blockIdx.z;
  const __bf16* wt  = (mode == 0) ? wtq : (mode == 1) ? wtk : wtv;
  const float* bias = (mode == 0) ? bq  : (mode == 1) ? bk  : bv;
  const int mBlk = blockIdx.x * 256;
  const int nBlk = blockIdx.y * 128;
  const int m0 = mBlk + wm * 64;   // token rows
  const int n0 = nBlk + wn * 64;   // hidden cols (one head: 64-aligned)
  const int ncol = lane & 15, hig = lane >> 4;

  v8f acc[4][4] = {};
  gemm_mainloop(xb, wt, mBlk, nBlk, As, Bs, tid, wm, wn, lane, acc);

  // bias epilogue
#pragma unroll
  for (int jn = 0; jn < 4; ++jn) {
    float bb = bias[n0 + jn * 16 + ncol];
#pragma unroll
    for (int jm = 0; jm < 4; ++jm)
#pragma unroll
      for (int r = 0; r < 8; ++r) acc[jm][jn][r] += bb;
  }
  const int h = n0 / DHEAD;
  if (mode < 2) {  // RoPE on Q and K; pairs (d, d+32) live in frags (jn, jn+2)
#pragma unroll
    for (int jn = 0; jn < 2; ++jn) {
      int dlo = jn * 16 + ncol;  // 0..31
      float invf = __powf(10000.0f, -(float)(2 * dlo) * (1.0f / (float)DHEAD));
#pragma unroll
      for (int jm = 0; jm < 4; ++jm)
#pragma unroll
        for (int r = 0; r < 8; ++r) {
          int tok = m0 + jm * 16 + r + 8 * hig;
          int pos = tok & (SEQ - 1);
          float ang = (float)pos * invf;
          float cs = __cosf(ang), sn = __sinf(ang);
          float lo = acc[jm][jn][r], hi2 = acc[jm][jn + 2][r];
          acc[jm][jn][r]     = lo * cs - hi2 * sn;
          acc[jm][jn + 2][r] = hi2 * cs + lo * sn;
        }
    }
  }
  // store: Q/K -> [B,H,T,D]; V -> [B,H,D,T] (transposed for PV B-fragments)
#pragma unroll
  for (int jm = 0; jm < 4; ++jm)
#pragma unroll
    for (int jn = 0; jn < 4; ++jn)
#pragma unroll
      for (int r = 0; r < 8; ++r) {
        int tok = m0 + jm * 16 + r + 8 * hig;
        int b = tok >> 11, tt = tok & (SEQ - 1);
        int d = jn * 16 + ncol;
        __bf16 val = f2bf(acc[jm][jn][r]);
        if (mode == 0)
          qout[(((size_t)(b * NHEAD + h)) * SEQ + tt) * DHEAD + d] = val;
        else if (mode == 1)
          kout[(((size_t)(b * NHEAD + h)) * SEQ + tt) * DHEAD + d] = val;
        else
          vtout[(((size_t)(b * NHEAD + h)) * DHEAD + d) * SEQ + tt] = val;
      }
}

// ---------------------------------------------------------------------------
// Flash attention, causal. One wave handles one 16-query tile of one (b,h).
// Block = 4 waves. Per-wave LDS tile re-lays P (C-layout) into A-fragments.
__global__ __launch_bounds__(128) void attn_kernel(const __bf16* __restrict__ qm,
                                                   const __bf16* __restrict__ km,
                                                   const __bf16* __restrict__ vt,
                                                   __bf16* __restrict__ aout) {
  __shared__ __bf16 plds[4][16 * 64];
  const int lane = threadIdx.x & 31;
  const int wv = threadIdx.x >> 5;
  const int gw = blockIdx.x * 4 + wv;
  const int qt = gw & 127;          // 128 q-tiles per (b,h)
  const int bh = gw >> 7;           // 0..63
  const __bf16* q = qm + (size_t)bh * SEQ * DHEAD;
  const __bf16* k = km + (size_t)bh * SEQ * DHEAD;
  const __bf16* v = vt + (size_t)bh * DHEAD * SEQ;
  const int q0 = qt * 16;
  const int ncol = lane & 15, hig = lane >> 4;

  v16bf qa0 = load_a(q, q0, DHEAD, 0, lane);
  v16bf qa1 = load_a(q, q0, DHEAD, 32, lane);

  v8f o[4] = {};
  float mrow[8], lrow[8];
#pragma unroll
  for (int r = 0; r < 8; ++r) { mrow[r] = -__builtin_inff(); lrow[r] = 0.0f; }

  const float scale = 0.125f;  // 1/sqrt(64)
  for (int kb = 0; kb < q0 + 16; kb += 64) {
    // S = Q @ K^T (16 x 64)
    v8f s[4] = {};
#pragma unroll
    for (int jn = 0; jn < 4; ++jn) {
      v16bf kb0 = load_b(k, kb + jn * 16, DHEAD, 0, lane);
      v16bf kb1 = load_b(k, kb + jn * 16, DHEAD, 32, lane);
      s[jn] = wmma_bf16(qa0, kb0, s[jn]);
      s[jn] = wmma_bf16(qa1, kb1, s[jn]);
    }
    // scale + causal mask + per-lane partial row max
    float pmax[8];
#pragma unroll
    for (int r = 0; r < 8; ++r) pmax[r] = -__builtin_inff();
#pragma unroll
    for (int jn = 0; jn < 4; ++jn)
#pragma unroll
      for (int r = 0; r < 8; ++r) {
        int qi = q0 + r + 8 * hig;
        int kj = kb + jn * 16 + ncol;
        float sv = (kj <= qi) ? s[jn][r] * scale : -__builtin_inff();
        s[jn][r] = sv;
        pmax[r] = fmaxf(pmax[r], sv);
      }
    // reduce over the 16 lanes sharing each row group
#pragma unroll
    for (int off = 1; off < 16; off <<= 1)
#pragma unroll
      for (int r = 0; r < 8; ++r) pmax[r] = fmaxf(pmax[r], __shfl_xor(pmax[r], off, 32));

    float alpha[8], psum[8];
#pragma unroll
    for (int r = 0; r < 8; ++r) {
      float mn = fmaxf(mrow[r], pmax[r]);
      alpha[r] = __expf(mrow[r] - mn);
      mrow[r] = mn;
      psum[r] = 0.0f;
    }
#pragma unroll
    for (int jn = 0; jn < 4; ++jn)
#pragma unroll
      for (int r = 0; r < 8; ++r) {
        float p = __expf(s[jn][r] - mrow[r]);
        s[jn][r] = p;
        psum[r] += p;
      }
#pragma unroll
    for (int off = 1; off < 16; off <<= 1)
#pragma unroll
      for (int r = 0; r < 8; ++r) psum[r] += __shfl_xor(psum[r], off, 32);
#pragma unroll
    for (int r = 0; r < 8; ++r) lrow[r] = lrow[r] * alpha[r] + psum[r];
#pragma unroll
    for (int jd = 0; jd < 4; ++jd)
#pragma unroll
      for (int r = 0; r < 8; ++r) o[jd][r] *= alpha[r];

    // P (C-layout) -> LDS -> A-fragments
    __bf16* pl = plds[wv];
#pragma unroll
    for (int jn = 0; jn < 4; ++jn)
#pragma unroll
      for (int r = 0; r < 8; ++r)
        pl[(r + 8 * hig) * 64 + jn * 16 + ncol] = f2bf(s[jn][r]);
    asm volatile("s_wait_dscnt 0x0" ::: "memory");  // wave-local DS RAW fence
    v16bf pa0 = load_a(pl, 0, 64, 0, lane);
    v16bf pa1 = load_a(pl, 0, 64, 32, lane);

    // O += P @ V  (V stored [d][t]: K-contiguous B fragments)
#pragma unroll
    for (int jd = 0; jd < 4; ++jd) {
      v16bf vb0 = load_b(v, jd * 16, SEQ, kb, lane);
      v16bf vb1 = load_b(v, jd * 16, SEQ, kb + 32, lane);
      o[jd] = wmma_bf16(pa0, vb0, o[jd]);
      o[jd] = wmma_bf16(pa1, vb1, o[jd]);
    }
  }
  // normalize + store to [B,T,HIDDEN] bf16 (token-major for output projection)
  const int h = bh & (NHEAD - 1), b = bh >> 4;
#pragma unroll
  for (int jd = 0; jd < 4; ++jd)
#pragma unroll
    for (int r = 0; r < 8; ++r) {
      int tt = q0 + r + 8 * hig;
      aout[((size_t)(b * SEQ + tt)) * HID + h * DHEAD + jd * 16 + ncol] =
          f2bf(o[jd][r] / lrow[r]);
    }
}

// ---------------------------------------------------------------------------
// Output projection: attn_bf [NTOK x HID] @ Wo + bo -> fp32 out (async-staged)
__global__ __launch_bounds__(256) void out_proj_kernel(const __bf16* __restrict__ ab,
                                                       const __bf16* __restrict__ wto,
                                                       const float* __restrict__ bo,
                                                       float* __restrict__ out) {
  __shared__ __bf16 As[2][256 * 64];
  __shared__ __bf16 Bs[2][128 * 64];
  const int tid = threadIdx.x;
  const int lane = tid & 31;
  const int wave = tid >> 5;
  const int wm = wave >> 1, wn = wave & 1;
  const int mBlk = blockIdx.x * 256;
  const int nBlk = blockIdx.y * 128;
  const int m0 = mBlk + wm * 64;
  const int n0 = nBlk + wn * 64;
  const int ncol = lane & 15, hig = lane >> 4;

  v8f acc[4][4] = {};
  gemm_mainloop(ab, wto, mBlk, nBlk, As, Bs, tid, wm, wn, lane, acc);

#pragma unroll
  for (int jn = 0; jn < 4; ++jn) {
    float bb = bo[n0 + jn * 16 + ncol];
#pragma unroll
    for (int jm = 0; jm < 4; ++jm)
#pragma unroll
      for (int r = 0; r < 8; ++r)
        out[((size_t)(m0 + jm * 16 + r + 8 * hig)) * HID + n0 + jn * 16 + ncol] =
            acc[jm][jn][r] + bb;
  }
}

// ---------------------------------------------------------------------------
extern "C" void kernel_launch(void* const* d_in, const int* in_sizes, int n_in,
                              void* d_out, int out_size, void* d_ws, size_t ws_size,
                              hipStream_t stream) {
  (void)in_sizes; (void)n_in; (void)out_size; (void)ws_size;
  const float* x  = (const float*)d_in[0];
  const float* Wq = (const float*)d_in[1];
  const float* bq = (const float*)d_in[2];
  const float* Wk = (const float*)d_in[3];
  const float* bk = (const float*)d_in[4];
  const float* Wv = (const float*)d_in[5];
  const float* bv = (const float*)d_in[6];
  const float* Wo = (const float*)d_in[7];
  const float* bo = (const float*)d_in[8];
  float* out = (float*)d_out;

  char* ws = (char*)d_ws;
  size_t off = 0;
  const size_t XB = (size_t)NTOK * HID * 2;
  const size_t WB = (size_t)HID * HID * 2;
  const size_t QB = (size_t)NBATCH * NHEAD * SEQ * DHEAD * 2;
  __bf16* xb  = (__bf16*)(ws + off); off += XB;
  __bf16* wtq = (__bf16*)(ws + off); off += WB;
  __bf16* wtk = (__bf16*)(ws + off); off += WB;
  __bf16* wtv = (__bf16*)(ws + off); off += WB;
  __bf16* wto = (__bf16*)(ws + off); off += WB;
  __bf16* qb  = (__bf16*)(ws + off); off += QB;
  __bf16* kb  = (__bf16*)(ws + off); off += QB;
  __bf16* vtb = (__bf16*)(ws + off); off += QB;
  __bf16* ab  = (__bf16*)(ws + off); off += XB;

  cvt_bf16_kernel<<<(NTOK * HID) / 256, 256, 0, stream>>>(x, xb, NTOK * HID);
  transpose_bf16_kernel<<<(HID * HID) / 256, 256, 0, stream>>>(Wq, wtq);
  transpose_bf16_kernel<<<(HID * HID) / 256, 256, 0, stream>>>(Wk, wtk);
  transpose_bf16_kernel<<<(HID * HID) / 256, 256, 0, stream>>>(Wv, wtv);
  transpose_bf16_kernel<<<(HID * HID) / 256, 256, 0, stream>>>(Wo, wto);

  qkv_rope_kernel<<<dim3(NTOK / 256, HID / 128, 3), 256, 0, stream>>>(
      xb, wtq, wtk, wtv, bq, bk, bv, qb, kb, vtb);

  attn_kernel<<<dim3((NBATCH * NHEAD * (SEQ / 16)) / 4), 128, 0, stream>>>(qb, kb, vtb,
                                                                           ab);

  out_proj_kernel<<<dim3(NTOK / 256, HID / 128), 256, 0, stream>>>(ab, wto, bo, out);
}
